// Attention_4329327034558
// MI455X (gfx1250) — matrix operand order repaired and checked by reading, not compiled, not run on hardware
//
#include <hip/hip_runtime.h>
#include <stdint.h>
#include <stddef.h>

// ---------------------------------------------------------------------------
// MI455X (gfx1250) attention block, bf16 WMMA (16x16x32) with f32 accumulate.
// Flash-attention style: sim ([32,2048,2048] = 512MB) is never materialized.
// Round 2: 32-row M-tiles everywhere so each B-fragment feeds two WMMAs
// (2x arithmetic intensity vs round 1).
// ---------------------------------------------------------------------------

typedef __bf16 bf16;
typedef __attribute__((ext_vector_type(8)))  bf16  v8bf;
typedef __attribute__((ext_vector_type(16))) bf16  v16bf;
typedef __attribute__((ext_vector_type(8)))  float v8f;

#define BATCH 4
#define HEADS 8
#define DHEAD 64
#define SEQ   2048
#define DIM   256
#define HID   512          // HEADS*DHEAD
#define QKV_ROWS 1536      // 3*HID
#define QSCALE 0.125f      // DHEAD^-0.5

// f32 -> bf16 (round to nearest even), bit-exact, no __bf16 arithmetic needed
__device__ __forceinline__ unsigned short f2bf(float f) {
  uint32_t u = __float_as_uint(f);
  uint32_t r = u + 0x7fffu + ((u >> 16) & 1u);
  return (unsigned short)(r >> 16);
}

__device__ __forceinline__ v8f vzero8f() {
  v8f v = {0.f, 0.f, 0.f, 0.f, 0.f, 0.f, 0.f, 0.f};
  return v;
}

// A fragment, 16x32 bf16, source row-major (row stride ld elems).
// CDNA5 layout: lane L holds row (L&15); elements e: cols = col_base +
// (L>>4)*8 + e  for e<8, and +16 for e>=8  -> two contiguous 16B chunks.
__device__ __forceinline__ v16bf load_a_frag(const unsigned short* src, int ld,
                                             int row_base, int col_base, int lane) {
  int sel = (lane >> 4) & 1;
  const unsigned short* p =
      src + (size_t)(row_base + (lane & 15)) * ld + col_base + sel * 8;
  v8bf lo = *(const v8bf*)p;
  v8bf hi = *(const v8bf*)(p + 16);
  return __builtin_shufflevector(lo, hi, 0, 1, 2, 3, 4, 5, 6, 7,
                                         8, 9, 10, 11, 12, 13, 14, 15);
}

// B fragment, 32x16 bf16, B[k][n] sourced from B-transposed storage
// srcT[n][k] (row stride ld). Lane L holds col n = n_base+(L&15);
// elements e: k = k_base + (L>>4)*8 + e for e<8, and +16 for e>=8
// -> two contiguous 16B chunks.
__device__ __forceinline__ v16bf load_b_frag(const unsigned short* srcT, int ld,
                                             int n_base, int k_base, int lane) {
  int sel = (lane >> 4) & 1;
  const unsigned short* p =
      srcT + (size_t)(n_base + (lane & 15)) * ld + k_base + sel * 8;
  v8bf lo = *(const v8bf*)p;
  v8bf hi = *(const v8bf*)(p + 16);
  return __builtin_shufflevector(lo, hi, 0, 1, 2, 3, 4, 5, 6, 7,
                                         8, 9, 10, 11, 12, 13, 14, 15);
}

__device__ __forceinline__ v8f wmma_bf16(v16bf a, v16bf b, v8f c) {
  return __builtin_amdgcn_wmma_f32_16x16x32_bf16(
      /*neg_a=*/false, a, /*neg_b=*/false, b,
      /*c_mod=*/(short)0, c, /*reuse_a=*/false, /*reuse_b=*/false);
}

// ---------------------------------------------------------------------------
// Prep kernels: convert to bf16 (and transpose x so GEMM B-fragments are
// contiguous 32B loads).
// ---------------------------------------------------------------------------
__global__ void k_prep_x(const float* __restrict__ x, unsigned short* __restrict__ xT) {
  int b = blockIdx.y;
  int idx = blockIdx.x * blockDim.x + threadIdx.x;   // [0, SEQ*DIM)
  int t = idx / DIM, i = idx % DIM;
  xT[((size_t)b * SEQ + t) * DIM + i] = f2bf(x[((size_t)b * DIM + i) * SEQ + t]);
}

__global__ void k_prep_w(const float* __restrict__ w, unsigned short* __restrict__ wb, int n) {
  int idx = blockIdx.x * blockDim.x + threadIdx.x;
  if (idx < n) wb[idx] = f2bf(w[idx]);
}

// ---------------------------------------------------------------------------
// QKV projection: [1536,256] x [256,2048] per batch. One wave = 32x64 tile
// (two 16-row A sub-tiles share every B fragment). Writes Q (scaled) and K
// as [b,h,n,d]; V transposed as [b,h,d,n] for the attention PV B-fragments.
// ---------------------------------------------------------------------------
__global__ void __launch_bounds__(32)
k_qkv(const unsigned short* __restrict__ wq, const unsigned short* __restrict__ xT,
      unsigned short* __restrict__ Q, unsigned short* __restrict__ K,
      unsigned short* __restrict__ VT) {
  int lane = threadIdx.x;
  int b = blockIdx.z;
  int m_base = blockIdx.y * 32;   // output channel tile (o)
  int n_base = blockIdx.x * 64;   // token tile
  const unsigned short* xTb = xT + (size_t)b * SEQ * DIM;

  v8f acc[2][4];
  for (int mt = 0; mt < 2; ++mt)
    for (int nt = 0; nt < 4; ++nt) acc[mt][nt] = vzero8f();

#pragma unroll
  for (int k = 0; k < DIM; k += 32) {
    v16bf a0 = load_a_frag(wq, DIM, m_base,      k, lane);
    v16bf a1 = load_a_frag(wq, DIM, m_base + 16, k, lane);
#pragma unroll
    for (int nt = 0; nt < 4; ++nt) {
      v16bf bb = load_b_frag(xTb, DIM, n_base + nt * 16, k, lane);
      acc[0][nt] = wmma_bf16(a0, bb, acc[0][nt]);
      acc[1][nt] = wmma_bf16(a1, bb, acc[1][nt]);
    }
  }

  int sel = lane >> 4, cl = lane & 15;
#pragma unroll
  for (int mt = 0; mt < 2; ++mt) {
    int seg = (m_base + mt * 16) >> 9;   // 0=Q, 1=K, 2=V (uniform per sub-tile)
#pragma unroll
    for (int nt = 0; nt < 4; ++nt) {
#pragma unroll
      for (int r = 0; r < 8; ++r) {
        int o = m_base + mt * 16 + sel * 8 + r;
        int t = n_base + nt * 16 + cl;
        float v = acc[mt][nt][r];
        int oo = o & 511;
        int h = oo >> 6, d = oo & 63;
        size_t bh = (size_t)b * HEADS + h;
        if (seg == 0)      Q [(bh * SEQ + t) * DHEAD + d] = f2bf(v * QSCALE);
        else if (seg == 1) K [(bh * SEQ + t) * DHEAD + d] = f2bf(v);
        else               VT[(bh * DHEAD + d) * SEQ + t] = f2bf(v);
      }
    }
  }
}

// ---------------------------------------------------------------------------
// Flash attention. One wave = 32 query rows (two 16-row sub-tiles); streams
// keys in chunks of 32. Per chunk: 4 K-fragments and 4 V-fragments are each
// consumed by two WMMAs (16 WMMA per chunk, 32 flop/byte of L2 traffic).
// Online softmax uses shfl_xor row reductions over the 16-lane fragment
// groups; P goes D-frag -> A-frag through a 2KB LDS transpose scratch.
// Output written bf16 as [b, t, hid] so the final GEMM's B-frags are
// contiguous.
// ---------------------------------------------------------------------------
__global__ void __launch_bounds__(32)
k_attn(const unsigned short* __restrict__ Q, const unsigned short* __restrict__ K,
       const unsigned short* __restrict__ VT, unsigned short* __restrict__ O) {
  __shared__ unsigned short lds[32 * 32];   // P tiles: 2 x (16 rows x 32 keys)
  int lane = threadIdx.x;
  int qt = blockIdx.x;                       // query tile (32 rows)
  int h = blockIdx.y, b = blockIdx.z;
  size_t bh = (size_t)b * HEADS + h;
  const unsigned short* Qh = Q + bh * SEQ * DHEAD;
  const unsigned short* Kh = K + bh * SEQ * DHEAD;
  const unsigned short* Vh = VT + bh * DHEAD * SEQ;
  int sel = lane >> 4, cl = lane & 15;

  // Q fragments loaded once: [sub-tile][d-halves]
  v16bf qa[2][2];
#pragma unroll
  for (int mt = 0; mt < 2; ++mt) {
    qa[mt][0] = load_a_frag(Qh, DHEAD, qt * 32 + mt * 16, 0,  lane);
    qa[mt][1] = load_a_frag(Qh, DHEAD, qt * 32 + mt * 16, 32, lane);
  }

  v8f o_acc[2][4];
  float m_r[2][8], l_r[2][8];
#pragma unroll
  for (int mt = 0; mt < 2; ++mt) {
    for (int dt = 0; dt < 4; ++dt) o_acc[mt][dt] = vzero8f();
#pragma unroll
    for (int r = 0; r < 8; ++r) { m_r[mt][r] = -3.0e38f; l_r[mt][r] = 0.0f; }
  }

  for (int kc = 0; kc < SEQ; kc += 32) {
    // K fragments for this chunk, shared by both query sub-tiles
    v16bf kb0 = load_b_frag(Kh, DHEAD, kc,      0,  lane);
    v16bf kb1 = load_b_frag(Kh, DHEAD, kc,      32, lane);
    v16bf kb2 = load_b_frag(Kh, DHEAD, kc + 16, 0,  lane);
    v16bf kb3 = load_b_frag(Kh, DHEAD, kc + 16, 32, lane);

#pragma unroll
    for (int mt = 0; mt < 2; ++mt) {
      // --- S = Q K^T for keys [kc, kc+32), rows [qt*32+mt*16, +16) ---
      v8f s0 = vzero8f(), s1 = vzero8f();
      s0 = wmma_bf16(qa[mt][0], kb0, s0);
      s0 = wmma_bf16(qa[mt][1], kb1, s0);
      s1 = wmma_bf16(qa[mt][0], kb2, s1);
      s1 = wmma_bf16(qa[mt][1], kb3, s1);
      // --- online softmax (row = sel*8 + r, spread across 16-lane group) ---
#pragma unroll
      for (int r = 0; r < 8; ++r) {
        float t = fmaxf(s0[r], s1[r]);
        t = fmaxf(t, __shfl_xor(t, 1));
        t = fmaxf(t, __shfl_xor(t, 2));
        t = fmaxf(t, __shfl_xor(t, 4));
        t = fmaxf(t, __shfl_xor(t, 8));
        float mnew  = fmaxf(m_r[mt][r], t);
        float alpha = __expf(m_r[mt][r] - mnew);
        float p0 = __expf(s0[r] - mnew);
        float p1 = __expf(s1[r] - mnew);
        float rs = p0 + p1;
        rs += __shfl_xor(rs, 1);
        rs += __shfl_xor(rs, 2);
        rs += __shfl_xor(rs, 4);
        rs += __shfl_xor(rs, 8);
        l_r[mt][r] = l_r[mt][r] * alpha + rs;
        m_r[mt][r] = mnew;
#pragma unroll
        for (int dt = 0; dt < 4; ++dt) o_acc[mt][dt][r] *= alpha;
        int row = sel * 8 + r;
        lds[(mt * 16 + row) * 32 + cl]      = f2bf(p0);
        lds[(mt * 16 + row) * 32 + 16 + cl] = f2bf(p1);
      }
    }
    __builtin_amdgcn_wave_barrier();   // keep DS store->load ordering (same wave, in-order LDS)
    // --- P fragments (A layout) from LDS ---
    v16bf pa0 = load_a_frag(lds,           32, 0, 0, lane);
    v16bf pa1 = load_a_frag(lds + 16 * 32, 32, 0, 0, lane);
    // --- O += P V : V fragments shared by both sub-tiles ---
#pragma unroll
    for (int dt = 0; dt < 4; ++dt) {
      v16bf vb = load_b_frag(Vh, SEQ, dt * 16, kc, lane);
      o_acc[0][dt] = wmma_bf16(pa0, vb, o_acc[0][dt]);
      o_acc[1][dt] = wmma_bf16(pa1, vb, o_acc[1][dt]);
    }
  }

  // epilogue: normalize and store as [b, t, hid] bf16
#pragma unroll
  for (int mt = 0; mt < 2; ++mt) {
#pragma unroll
    for (int dt = 0; dt < 4; ++dt) {
#pragma unroll
      for (int r = 0; r < 8; ++r) {
        float v = o_acc[mt][dt][r] / l_r[mt][r];
        int t  = qt * 32 + mt * 16 + sel * 8 + r;
        int hd = h * DHEAD + dt * 16 + cl;
        O[((size_t)b * SEQ + t) * HID + hd] = f2bf(v);
      }
    }
  }
}

// ---------------------------------------------------------------------------
// Output projection: out[b,o,t] = sum_hd w_out[o,hd]*O[b,t,hd] + b_out[o]
// One wave = 32x64 tile, f32 output.
// ---------------------------------------------------------------------------
__global__ void __launch_bounds__(32)
k_proj(const unsigned short* __restrict__ wo, const unsigned short* __restrict__ O,
       const float* __restrict__ b_out, float* __restrict__ out) {
  int lane = threadIdx.x;
  int b = blockIdx.z;
  int m_base = blockIdx.y * 32;   // output channel tile
  int n_base = blockIdx.x * 64;   // token tile
  const unsigned short* Ob = O + (size_t)b * SEQ * HID;

  v8f acc[2][4];
  for (int mt = 0; mt < 2; ++mt)
    for (int nt = 0; nt < 4; ++nt) acc[mt][nt] = vzero8f();

  for (int k = 0; k < HID; k += 32) {
    v16bf a0 = load_a_frag(wo, HID, m_base,      k, lane);
    v16bf a1 = load_a_frag(wo, HID, m_base + 16, k, lane);
#pragma unroll
    for (int nt = 0; nt < 4; ++nt) {
      v16bf bb = load_b_frag(Ob, HID, n_base + nt * 16, k, lane);
      acc[0][nt] = wmma_bf16(a0, bb, acc[0][nt]);
      acc[1][nt] = wmma_bf16(a1, bb, acc[1][nt]);
    }
  }

  int sel = lane >> 4, cl = lane & 15;
#pragma unroll
  for (int mt = 0; mt < 2; ++mt) {
#pragma unroll
    for (int nt = 0; nt < 4; ++nt) {
#pragma unroll
      for (int r = 0; r < 8; ++r) {
        int o = m_base + mt * 16 + sel * 8 + r;
        int t = n_base + nt * 16 + cl;
        out[((size_t)b * DIM + o) * SEQ + t] = acc[mt][nt][r] + b_out[o];
      }
    }
  }
}

// ---------------------------------------------------------------------------
extern "C" void kernel_launch(void* const* d_in, const int* in_sizes, int n_in,
                              void* d_out, int out_size, void* d_ws, size_t ws_size,
                              hipStream_t stream) {
  const float* x     = (const float*)d_in[0];   // [4,256,2048]
  const float* w_qkv = (const float*)d_in[1];   // [1536,256]
  const float* w_out = (const float*)d_in[2];   // [256,512]
  const float* b_out = (const float*)d_in[3];   // [256]
  float* out = (float*)d_out;                   // [4,256,2048]

  // Workspace carve-out (~37 MiB total), 256B aligned segments.
  char* ws = (char*)d_ws;
  size_t off = 0;
  auto carve = [&](size_t bytes) -> char* {
    char* p = ws + off;
    off = (off + bytes + 255) & ~(size_t)255;
    return p;
  };
  unsigned short* XT = (unsigned short*)carve((size_t)BATCH * SEQ * DIM * 2);       // x^T bf16
  unsigned short* WQ = (unsigned short*)carve((size_t)QKV_ROWS * DIM * 2);          // w_qkv bf16
  unsigned short* WO = (unsigned short*)carve((size_t)DIM * HID * 2);               // w_out bf16
  unsigned short* Qb = (unsigned short*)carve((size_t)BATCH * HID * SEQ * 2);       // [b,h,n,d]
  unsigned short* Kb = (unsigned short*)carve((size_t)BATCH * HID * SEQ * 2);       // [b,h,n,d]
  unsigned short* VT = (unsigned short*)carve((size_t)BATCH * HID * SEQ * 2);       // [b,h,d,n]
  unsigned short* OA = (unsigned short*)carve((size_t)BATCH * SEQ * HID * 2);       // [b,t,hid]
  (void)ws_size; (void)in_sizes; (void)n_in; (void)out_size;

  k_prep_x<<<dim3((SEQ * DIM) / 256, BATCH), 256, 0, stream>>>(x, XT);
  k_prep_w<<<(QKV_ROWS * DIM) / 256, 256, 0, stream>>>(w_qkv, WQ, QKV_ROWS * DIM);
  k_prep_w<<<(DIM * HID) / 256, 256, 0, stream>>>(w_out, WO, DIM * HID);

  k_qkv<<<dim3(SEQ / 64, QKV_ROWS / 32, BATCH), 32, 0, stream>>>(WQ, XT, Qb, Kb, VT);
  k_attn<<<dim3(SEQ / 32, HEADS, BATCH), 32, 0, stream>>>(Qb, Kb, VT, OA);
  k_proj<<<dim3(SEQ / 64, DIM / 32, BATCH), 32, 0, stream>>>(WO, OA, b_out, out);
}
